// ESA_20169166422866
// MI455X (gfx1250) — matrix operand order repaired and checked by reading, not compiled, not run on hardware
//
#include <hip/hip_runtime.h>
#include <hip/hip_bf16.h>
#include <math.h>
#include <stdint.h>

typedef __attribute__((ext_vector_type(16))) _Float16 v16h;
typedef __attribute__((ext_vector_type(8)))  float    v8f;

#define PIX 16384   // 128*128
#define BATCH 4

// ---------------------------------------------------------------------------
// WMMA fragment helpers (A/B in this code's contiguous K order, wave32)
// A 16x32 f16: lane l holds row M=l&15; halves e=0..15 -> K = (l>>4)*16 + e
// B stored transposed in LDS as [N][K]: lane l holds col N=l&15;
//              halves e=0..15 -> K = (l>>4)*16 + e
// C/D f32: reg r, lane l -> (M = r + (l>=16?8:0), N = l&15)
// ---------------------------------------------------------------------------
__device__ __forceinline__ v16h frag_a_lds(const _Float16* tile, int row0, int ldk, int k0) {
    int l = threadIdx.x & 31;
    const _Float16* p = tile + (size_t)(row0 + (l & 15)) * ldk + k0 + ((l >> 4) << 4);
    v16h r;
#pragma unroll
    for (int i = 0; i < 16; ++i) r[i] = p[i];
    return r;
}

__device__ __forceinline__ v16h frag_b_lds(const _Float16* tile, int row0, int ldk, int k0) {
    int l = threadIdx.x & 31;
    const _Float16* p = tile + (size_t)(row0 + (l & 15)) * ldk + k0 + ((l >> 4) << 4);
    v16h r;
#pragma unroll
    for (int i = 0; i < 16; ++i) r[i] = p[i];
    return r;
}

__device__ __forceinline__ v8f wmma16x16x32(v16h a, v16h b, v8f c) {
    return __builtin_amdgcn_wmma_f32_16x16x32_f16(false, a, false, b, (short)0, c, false, false);
}

__device__ __forceinline__ float gelu_f(float x) {
    return 0.5f * x * (1.0f + erff(x * 0.70710678118654752440f));
}

// async global->LDS bulk copy, 16B per lane-op, tracked on ASYNCcnt
__device__ __forceinline__ void async_copy_b128(uint32_t lds_off, const void* gaddr) {
    asm volatile("global_load_async_to_lds_b128 %0, %1, off"
                 :: "v"(lds_off), "v"(gaddr) : "memory");
}
__device__ __forceinline__ void wait_asynccnt0() {
    asm volatile("s_wait_asynccnt 0" ::: "memory");
}

// ---------------------------------------------------------------------------
// 0) convert an f32 weight matrix [M][K] to padded f16 [M][LDK] (zero pad)
// ---------------------------------------------------------------------------
__global__ __launch_bounds__(256) void prep_w_k(const float* __restrict__ src,
                                                _Float16* __restrict__ dst,
                                                int M, int K, int LDK) {
    int i = blockIdx.x * 256 + threadIdx.x;
    if (i >= M * LDK) return;
    int m = i / LDK, c = i % LDK;
    dst[i] = (_Float16)((c < K) ? src[m * K + c] : 0.f);
}

// ---------------------------------------------------------------------------
// 1) 2x2 max pool  (B,64,256,256) -> (B,64,128,128)
// ---------------------------------------------------------------------------
__global__ __launch_bounds__(256) void maxpool_k(const float* __restrict__ X,
                                                 float* __restrict__ Y) {
    int i = blockIdx.x * 256 + threadIdx.x;
    if (i >= BATCH * 64 * PIX) return;
    int xx = i & 127, yy = (i >> 7) & 127, c = (i >> 14) & 63, b = i >> 20;
    const float* base = X + ((size_t)b * 64 + c) * 65536;
    int o = (2 * yy) * 256 + 2 * xx;
    float v0 = base[o], v1 = base[o + 1], v2 = base[o + 256], v3 = base[o + 257];
    Y[i] = fmaxf(fmaxf(v0, v1), fmaxf(v2, v3));
}

// ---------------------------------------------------------------------------
// 2) generic 1x1-conv GEMM via WMMA: Y[b][m][p] = sum_k W[m][k]*X[b][k][p]
//    Weights arrive pre-converted f16 padded [M][LDK]; staged to LDS with
//    GLOBAL_LOAD_ASYNC_TO_LDS_B128 (overlaps the f32->f16 transposed X stage).
//    EPI: 0 none | 1 +bias | 2 gelu(y+bias)*E0 | 3 y+bias+E0
// ---------------------------------------------------------------------------
template <int M, int K, int EPI>
__global__ __launch_bounds__(128) void conv1x1_wmma(const float* __restrict__ X,
                                                    const _Float16* __restrict__ Wf,
                                                    const float* __restrict__ bias,
                                                    const float* __restrict__ E0,
                                                    float* __restrict__ Y, int P) {
    constexpr int LDK = K + 8;
    constexpr int WBYTES = M * LDK * 2;      // multiple of 16
    __shared__ _Float16 sW[M * LDK];
    __shared__ _Float16 sX[64 * LDK];
    const int b = blockIdx.y;
    const int p0 = blockIdx.x * 64;
    const int tid = threadIdx.x;

    // async weight copy (contiguous, already padded/converted in global)
    {
        uint32_t lds_base = (uint32_t)(uintptr_t)(&sW[0]);
        const char* g = (const char*)Wf;
        for (int off = tid * 16; off < WBYTES; off += 128 * 16)
            async_copy_b128(lds_base + off, g + off);
    }
    // transposed + converted activation tile (VALU path, overlaps async copy)
    for (int j = tid; j < 64 * K; j += 128) {
        int px = j & 63, c = j >> 6;
        sX[px * LDK + c] = (_Float16)X[((size_t)b * K + c) * P + p0 + px];
    }
    wait_asynccnt0();
    __syncthreads();

    const int wave = tid >> 5;
    const int l = tid & 31;
    const int hi = (l >> 4) << 3;
    const int ln = l & 15;
    v8f zero = {};

    for (int mt = wave; mt < M / 16; mt += 4) {
        v8f acc[4] = {zero, zero, zero, zero};
#pragma unroll
        for (int ks = 0; ks < K / 32; ++ks) {
            v16h a = frag_a_lds(sW, mt * 16, LDK, ks * 32);
#pragma unroll
            for (int nt = 0; nt < 4; ++nt) {
                v16h bb = frag_b_lds(sX, nt * 16, LDK, ks * 32);
                acc[nt] = wmma16x16x32(a, bb, acc[nt]);
            }
        }
#pragma unroll
        for (int nt = 0; nt < 4; ++nt) {
#pragma unroll
            for (int r = 0; r < 8; ++r) {
                int m = mt * 16 + r + hi;
                int px = p0 + nt * 16 + ln;
                float y = acc[nt][r];
                if (EPI == 1) {
                    y += bias[m];
                } else if (EPI == 2) {
                    float mv = E0[((size_t)b * M + m) * P + px];
                    y = gelu_f(y + bias[m]) * mv;
                } else if (EPI == 3) {
                    y += bias[m] + E0[((size_t)b * M + m) * P + px];
                }
                Y[((size_t)b * M + m) * P + px] = y;
            }
        }
    }
}

// ---------------------------------------------------------------------------
// 3) depthwise 3x3, zero pad (C channels)
// ---------------------------------------------------------------------------
__global__ __launch_bounds__(256) void dw3x3_zero_k(const float* __restrict__ X,
                                                    const float* __restrict__ W,
                                                    float* __restrict__ Y, int C) {
    int i = blockIdx.x * 256 + threadIdx.x;
    if (i >= BATCH * C * PIX) return;
    int xx = i & 127, yy = (i >> 7) & 127;
    int c = (i >> 14) % C;
    int b = i / (C * PIX);
    const float* base = X + ((size_t)b * C + c) * PIX;
    const float* w = W + c * 9;
    float s = 0.f;
#pragma unroll
    for (int dy = -1; dy <= 1; ++dy)
#pragma unroll
        for (int dx = -1; dx <= 1; ++dx) {
            int py = yy + dy, px = xx + dx;
            if (py >= 0 && py < 128 && px >= 0 && px < 128)
                s += w[(dy + 1) * 3 + (dx + 1)] * base[py * 128 + px];
        }
    Y[i] = s;
}

// depthwise 3x3, reflect pad, + bias (64 channels)
__global__ __launch_bounds__(256) void dw3x3_reflect_k(const float* __restrict__ X,
                                                       const float* __restrict__ W,
                                                       const float* __restrict__ bi,
                                                       float* __restrict__ Y) {
    int i = blockIdx.x * 256 + threadIdx.x;
    if (i >= BATCH * 64 * PIX) return;
    int xx = i & 127, yy = (i >> 7) & 127, c = (i >> 14) & 63, b = i >> 20;
    const float* base = X + ((size_t)b * 64 + c) * PIX;
    const float* w = W + c * 9;
    float s = 0.f;
#pragma unroll
    for (int dy = -1; dy <= 1; ++dy)
#pragma unroll
        for (int dx = -1; dx <= 1; ++dx) {
            int py = yy + dy, px = xx + dx;
            py = (py < 0) ? -py : ((py > 127) ? 254 - py : py);
            px = (px < 0) ? -px : ((px > 127) ? 254 - px : px);
            s += w[(dy + 1) * 3 + (dx + 1)] * base[py * 128 + px];
        }
    Y[i] = s + bi[c];
}

// ---------------------------------------------------------------------------
// 4) channel covariance via WMMA: cov = f . f^T (A and B^T share one LDS tile)
//    deterministic two-stage reduction, no atomics.
//    Each wave owns tiles (wave) and (wave+8) with NAMED accumulators so the
//    compiler never emits movrel-indexed VGPR arrays around the WMMAs.
// ---------------------------------------------------------------------------
__global__ __launch_bounds__(256) void cov_stage1(const float* __restrict__ qkv,
                                                  float* __restrict__ part,
                                                  float* __restrict__ psum, int P) {
    constexpr int LDK = 128 + 8;
    __shared__ _Float16 sF[64 * LDK];
    int b = blockIdx.y, ch = blockIdx.x;   // 32 chunks of 512 pixels
    int tid = threadIdx.x;
    int wave = tid >> 5;
    const int mt0 = wave >> 2, nt0 = wave & 3;            // tile id = wave
    const int mt1 = (wave + 8) >> 2, nt1 = (wave + 8) & 3; // tile id = wave+8
    v8f zero = {};
    v8f acc0 = zero, acc1 = zero;
    float rsum = 0.f;
    for (int sub = 0; sub < 4; ++sub) {
        int p0 = ch * 512 + sub * 128;
        for (int j = tid; j < 64 * 128; j += 256) {
            int px = j & 127, c = j >> 7;
            sF[c * LDK + px] = (_Float16)qkv[((size_t)b * 192 + c) * P + p0 + px];
        }
        __syncthreads();
        if (tid < 64) {  // channel sums (ordering-only use, f16 rounding OK)
            float s = 0.f;
            for (int px = 0; px < 128; ++px) s += (float)sF[tid * LDK + px];
            rsum += s;
        }
#pragma unroll
        for (int ks = 0; ks < 4; ++ks) {
            v16h a  = frag_a_lds(sF, mt0 * 16, LDK, ks * 32);
            v16h bb = frag_b_lds(sF, nt0 * 16, LDK, ks * 32);
            acc0 = wmma16x16x32(a, bb, acc0);
        }
#pragma unroll
        for (int ks = 0; ks < 4; ++ks) {
            v16h a  = frag_a_lds(sF, mt1 * 16, LDK, ks * 32);
            v16h bb = frag_b_lds(sF, nt1 * 16, LDK, ks * 32);
            acc1 = wmma16x16x32(a, bb, acc1);
        }
        __syncthreads();
    }
    int l = tid & 31, hi = (l >> 4) << 3, ln = l & 15;
    float* pbase = part + ((size_t)(b * 32 + ch)) * 4096;
#pragma unroll
    for (int r = 0; r < 8; ++r) {
        int c = mt0 * 16 + r + hi, d = nt0 * 16 + ln;
        pbase[c * 64 + d] = acc0[r];
    }
#pragma unroll
    for (int r = 0; r < 8; ++r) {
        int c = mt1 * 16 + r + hi, d = nt1 * 16 + ln;
        pbase[c * 64 + d] = acc1[r];
    }
    if (tid < 64) psum[(b * 32 + ch) * 64 + tid] = rsum;
}

__global__ __launch_bounds__(256) void cov_finish(const float* __restrict__ part,
                                                  const float* __restrict__ psum,
                                                  int* __restrict__ idxOut, int P) {
    __shared__ float cov[4096];
    __shared__ float S[64];
    __shared__ float stdv[64];
    __shared__ float sim[64];
    int b = blockIdx.x, tid = threadIdx.x;
    if (tid < 64) {
        float s = 0.f;
        for (int ch = 0; ch < 32; ++ch) s += psum[(b * 32 + ch) * 64 + tid];
        S[tid] = s;
    }
    for (int pr = tid; pr < 4096; pr += 256) {
        float s = 0.f;
        for (int ch = 0; ch < 32; ++ch) s += part[((size_t)(b * 32 + ch)) * 4096 + pr];
        cov[pr] = s;
    }
    __syncthreads();
    float invN = 1.0f / (float)P;
    if (tid < 64)
        stdv[tid] = sqrtf(cov[tid * 64 + tid] - S[tid] * S[tid] * invN + 1e-8f);
    __syncthreads();
    if (tid < 64) {
        float s = 0.f;
        for (int d = 0; d < 64; ++d) {
            float cd = cov[tid * 64 + d] - S[tid] * S[d] * invN;
            s += cd / fmaxf(stdv[tid] * stdv[d], 1e-8f);
        }
        sim[tid] = s * (1.0f / 64.0f);
    }
    __syncthreads();
    if (tid < 64) {
        float mv = sim[tid];
        int rank = 0;
        for (int d = 0; d < 64; ++d)
            rank += (sim[d] > mv) || (sim[d] == mv && d < tid);
        idxOut[b * 64 + rank] = tid;   // idx[rank] = channel
    }
}

// ---------------------------------------------------------------------------
// 5) halo attention + gate, one wave per (spatial block, head).
//    Writes gated output scattered to UNSORTED channels (out[idx[pos]] = ...).
// ---------------------------------------------------------------------------
__global__ __launch_bounds__(32) void attn_k(const float* __restrict__ qkv,
                                             const int* __restrict__ idx,
                                             const float* __restrict__ gw,
                                             const float* __restrict__ gb,
                                             const float* __restrict__ temp,
                                             const float* __restrict__ relh,
                                             const float* __restrict__ relw,
                                             float* __restrict__ out) {
    __shared__ _Float16 sQ[64][32];     // [q-pixel][ch] raw (K pad 32)
    __shared__ float    sQs[64];        // l2norm scale * exp(temp)
    __shared__ _Float16 sK[112][32];    // [k-pos][ch] (+rel emb), pad rows zero
    __shared__ float    sKs[112];
    __shared__ _Float16 sV[16][128];    // [ch][k-pos] (K pad 128)
    __shared__ _Float16 sGW[16][32];    // gate_w tile
    __shared__ _Float16 sA[16][128];    // attn weights (A frag for AV)
    __shared__ float    sL[16][112];    // logits / gate transpose scratch
    __shared__ int      sCh[16];

    const int wg = blockIdx.x;
    const int head = wg & 3;
    const int blk = wg >> 2;
    const int by = blk >> 4, bx = blk & 15;
    const int b = blockIdx.y;
    const int tid = threadIdx.x;

    if (tid < 16) sCh[tid] = idx[b * 64 + head * 16 + tid];
    __syncthreads();

    // Q (64 px x 16 ch)
    for (int j = tid; j < 64 * 16; j += 32) {
        int px = j >> 4, c = j & 15;
        int row = by * 8 + (px >> 3), col = bx * 8 + (px & 7);
        float v = qkv[(((size_t)b * 192 + sCh[c]) * 128 + row) * 128 + col];
        sQ[px][c] = (_Float16)v;
        sQ[px][c + 16] = (_Float16)0.f;
    }
    // K halo (100 kpos x 16 ch) + rel embeddings, rows 100..111 zero
    for (int j = tid; j < 112 * 16; j += 32) {
        int kp = j >> 4, c = j & 15;
        float v = 0.f;
        if (kp < 100) {
            int r = kp / 10, cc = kp % 10;
            int gr = by * 8 - 1 + r, gc = bx * 8 - 1 + cc;
            float base = (c < 8) ? relh[(head * 10 + r) * 8 + c]
                                 : relw[(head * 10 + cc) * 8 + (c - 8)];
            float kv = 0.f;
            if (gr >= 0 && gr < 128 && gc >= 0 && gc < 128)
                kv = qkv[(((size_t)b * 192 + 64 + sCh[c]) * 128 + gr) * 128 + gc];
            v = kv + base;
        }
        sK[kp][c] = (_Float16)v;
        sK[kp][c + 16] = (_Float16)0.f;
    }
    // V halo transposed: [ch][kpos], pad kpos 100..127 zero
    for (int j = tid; j < 16 * 128; j += 32) {
        int c = j >> 7, kp = j & 127;
        float v = 0.f;
        if (kp < 100) {
            int r = kp / 10, cc = kp % 10;
            int gr = by * 8 - 1 + r, gc = bx * 8 - 1 + cc;
            if (gr >= 0 && gr < 128 && gc >= 0 && gc < 128)
                v = qkv[(((size_t)b * 192 + 128 + sCh[c]) * 128 + gr) * 128 + gc];
        }
        sV[c][kp] = (_Float16)v;
    }
    // gate weights
    for (int j = tid; j < 16 * 16; j += 32) {
        int o = j >> 4, c = j & 15;
        sGW[o][c] = (_Float16)gw[(head * 16 + o) * 16 + c];
        sGW[o][c + 16] = (_Float16)0.f;
    }
    __syncthreads();

    const float tsc = __expf(temp[head]);
    for (int px = tid; px < 64; px += 32) {
        float s = 0.f;
        for (int c = 0; c < 16; ++c) { float v = (float)sQ[px][c]; s += v * v; }
        sQs[px] = tsc / fmaxf(sqrtf(s), 1e-12f);
    }
    for (int kp = tid; kp < 112; kp += 32) {
        float s = 0.f;
        for (int c = 0; c < 16; ++c) { float v = (float)sK[kp][c]; s += v * v; }
        sKs[kp] = (kp < 100) ? 1.0f / fmaxf(sqrtf(s), 1e-12f) : 0.0f;
    }
    __syncthreads();

    const int l = tid;
    const int hi = (l >> 4) << 3;
    const int ln = l & 15;
    v8f zero = {};

    for (int mt = 0; mt < 4; ++mt) {
        // ---- logits = l2norm(q) . l2norm(k)^T * exp(temp) ----
        {
            float qs = sQs[mt * 16 + ln];
            v16h a = frag_a_lds(&sQ[0][0], mt * 16, 32, 0);
#pragma unroll
            for (int i = 0; i < 16; ++i) a[i] = (_Float16)((float)a[i] * qs);
            for (int nt = 0; nt < 7; ++nt) {
                float ksc = sKs[nt * 16 + ln];
                v16h bb = frag_b_lds(&sK[0][0], nt * 16, 32, 0);
#pragma unroll
                for (int i = 0; i < 16; ++i) bb[i] = (_Float16)((float)bb[i] * ksc);
                v8f acc = wmma16x16x32(a, bb, zero);
#pragma unroll
                for (int r = 0; r < 8; ++r) sL[r + hi][nt * 16 + ln] = acc[r];
            }
        }
        __syncthreads();
        // ---- softmax over 100 valid k positions ----
        if (tid < 16) {
            float mx = -1e30f;
            for (int kp = 0; kp < 100; ++kp) mx = fmaxf(mx, sL[tid][kp]);
            float sum = 0.f;
            for (int kp = 0; kp < 100; ++kp) sum += __expf(sL[tid][kp] - mx);
            float inv = 1.0f / sum;
            for (int kp = 0; kp < 100; ++kp)
                sA[tid][kp] = (_Float16)(__expf(sL[tid][kp] - mx) * inv);
        } else {
            int row = tid - 16;
            for (int kp = 100; kp < 128; ++kp) sA[row][kp] = (_Float16)0.f;
        }
        __syncthreads();
        // ---- out = attn @ V   (K = 128 -> 4 wmma k-steps) ----
        v8f oacc = zero;
#pragma unroll
        for (int ks = 0; ks < 4; ++ks) {
            v16h a2 = frag_a_lds(&sA[0][0], 0, 128, ks * 32);
            v16h b2 = frag_b_lds(&sV[0][0], 0, 128, ks * 32);
            oacc = wmma16x16x32(a2, b2, oacc);
        }
        // ---- gate = gate_w @ q_raw (transposed vs out -> exchange via sL) ----
        {
            v16h ga = frag_a_lds(&sGW[0][0], 0, 32, 0);
            v16h gq = frag_b_lds(&sQ[0][0], mt * 16, 32, 0);
            v8f gacc = wmma16x16x32(ga, gq, zero);
#pragma unroll
            for (int r = 0; r < 8; ++r) {
                int o = r + hi;
                sL[o][ln] = gacc[r] + gb[head * 16 + o];
            }
        }
        __syncthreads();
#pragma unroll
        for (int r = 0; r < 8; ++r) {
            int pxl = r + hi;               // local pixel index in m-tile
            int px = mt * 16 + pxl;
            float gval = sL[ln][pxl];       // gate for (ch=ln, pixel=pxl)
            float sg = 1.0f / (1.0f + __expf(-gval));
            float v = oacc[r] * sg;
            int row = by * 8 + (px >> 3), col = bx * 8 + (px & 7);
            out[(((size_t)b * 64 + sCh[ln]) * 128 + row) * 128 + col] = v;
        }
        __syncthreads();
    }
}

// ---------------------------------------------------------------------------
// 6) mixed = attn_out(unsorted) + q + k   (cache unsorts to q+k exactly)
// ---------------------------------------------------------------------------
__global__ __launch_bounds__(256) void mixed_k(const float* __restrict__ attn,
                                               const float* __restrict__ qkv,
                                               float* __restrict__ Y) {
    int i = blockIdx.x * 256 + threadIdx.x;
    if (i >= BATCH * 64 * PIX) return;
    int p = i & (PIX - 1);
    int c = (i >> 14) & 63;
    int b = i >> 20;
    float q = qkv[((size_t)b * 192 + c) * PIX + p];
    float k = qkv[((size_t)b * 192 + 64 + c) * PIX + p];
    Y[i] = attn[i] + q + k;
}

// ---------------------------------------------------------------------------
// 7) bilinear x2 upsample with reference's (n_in-1)/(n_out-1) grid
// ---------------------------------------------------------------------------
__global__ __launch_bounds__(256) void upsample_k(const float* __restrict__ X,
                                                  float* __restrict__ Y) {
    int i = blockIdx.x * 256 + threadIdx.x;
    if (i >= BATCH * 64 * 256 * 256) return;
    int ox = i & 255, oy = (i >> 8) & 255, c = (i >> 16) & 63, b = i >> 22;
    const float sc = 127.0f / 255.0f;
    float cy = oy * sc;
    int ly = (int)floorf(cy); ly = ly > 126 ? 126 : ly;
    float wy = cy - (float)ly;
    float cx = ox * sc;
    int lx = (int)floorf(cx); lx = lx > 126 ? 126 : lx;
    float wx = cx - (float)lx;
    const float* base = X + ((size_t)b * 64 + c) * PIX;
    float v00 = base[ly * 128 + lx],       v01 = base[ly * 128 + lx + 1];
    float v10 = base[(ly + 1) * 128 + lx], v11 = base[(ly + 1) * 128 + lx + 1];
    Y[i] = v00 * (1.f - wy) * (1.f - wx) + v01 * (1.f - wy) * wx +
           v10 * wy * (1.f - wx) + v11 * wy * wx;
}

// ---------------------------------------------------------------------------
extern "C" void kernel_launch(void* const* d_in, const int* in_sizes, int n_in,
                              void* d_out, int out_size, void* d_ws, size_t ws_size,
                              hipStream_t stream) {
    (void)in_sizes; (void)n_in; (void)out_size; (void)ws_size;
    const float* x      = (const float*)d_in[0];
    const float* qkv_w  = (const float*)d_in[1];
    const float* lce_w  = (const float*)d_in[2];
    const float* gate_w = (const float*)d_in[3];
    const float* gate_b = (const float*)d_in[4];
    const float* temp   = (const float*)d_in[5];
    const float* relh   = (const float*)d_in[6];
    const float* relw   = (const float*)d_in[7];
    const float* down_w = (const float*)d_in[8];
    const float* down_b = (const float*)d_in[9];
    const float* up_w   = (const float*)d_in[10];
    const float* up_b   = (const float*)d_in[11];
    const float* gat_w  = (const float*)d_in[12];
    const float* gat_b  = (const float*)d_in[13];
    const float* proj_w = (const float*)d_in[14];
    const float* lp_w   = (const float*)d_in[15];
    const float* lp_b   = (const float*)d_in[16];

    float* ws = (float*)d_ws;
    const int P = PIX;
    // workspace layout (floats); live ranges are disjoint for aliased buffers
    float* xp    = ws;                 // [B][64][P]   pooled input
    float* qpre  = ws + 4194304;       // [B][192][P]  conv1x1 qkv (pre-dw)
    float* qkvb  = ws + 16777216;      // [B][192][P]  after depthwise
    float* attnb = ws + 29360128;      // [B][64][P]   gated attention (unsorted)
    float* mixedb= qpre;               // reuse qpre region
    float* gbuf  = ws + 8388608;
    float* tbuf  = ws + 12582912;      // [B][32][P]
    float* ubuf  = xp;                 // reuse xp
    float* prbuf = qkvb;               // reuse qkv region
    float* lpbuf = attnb;              // reuse attn region
    float* part  = ws + 33554432;      // [B][32][4096]
    float* psum  = ws + 34078720;      // [B][32][64]
    int*   idxb  = (int*)(ws + 34086912); // [B][64]
    // f16 padded weights, all 16B-aligned slots
    _Float16* wqkv  = (_Float16*)(ws + 34087168); // 192*72
    _Float16* wgat  = (_Float16*)(ws + 34094080); // 64*72
    _Float16* wdown = (_Float16*)(ws + 34096384); // 32*72
    _Float16* wup   = (_Float16*)(ws + 34097536); // 64*40
    _Float16* wproj = (_Float16*)(ws + 34098816); // 64*72

    // weight conversion (tiny, once per launch)
    prep_w_k<<<dim3((192 * 72 + 255) / 256), 256, 0, stream>>>(qkv_w, wqkv, 192, 64, 72);
    prep_w_k<<<dim3((64 * 72 + 255) / 256), 256, 0, stream>>>(gat_w, wgat, 64, 64, 72);
    prep_w_k<<<dim3((32 * 72 + 255) / 256), 256, 0, stream>>>(down_w, wdown, 32, 64, 72);
    prep_w_k<<<dim3((64 * 40 + 255) / 256), 256, 0, stream>>>(up_w, wup, 64, 32, 40);
    prep_w_k<<<dim3((64 * 72 + 255) / 256), 256, 0, stream>>>(proj_w, wproj, 64, 64, 72);

    maxpool_k<<<dim3(BATCH * 64 * PIX / 256), 256, 0, stream>>>(x, xp);
    conv1x1_wmma<192, 64, 0><<<dim3(P / 64, BATCH), 128, 0, stream>>>(
        xp, wqkv, nullptr, nullptr, qpre, P);
    dw3x3_zero_k<<<dim3(BATCH * 192 * PIX / 256), 256, 0, stream>>>(qpre, lce_w, qkvb, 192);
    cov_stage1<<<dim3(32, BATCH), 256, 0, stream>>>(qkvb, part, psum, P);
    cov_finish<<<dim3(BATCH), 256, 0, stream>>>(part, psum, idxb, P);
    attn_k<<<dim3(1024, BATCH), 32, 0, stream>>>(qkvb, idxb, gate_w, gate_b, temp,
                                                 relh, relw, attnb);
    mixed_k<<<dim3(BATCH * 64 * PIX / 256), 256, 0, stream>>>(attnb, qkvb, mixedb);
    conv1x1_wmma<64, 64, 2><<<dim3(P / 64, BATCH), 128, 0, stream>>>(
        mixedb, wgat, gat_b, mixedb, gbuf, P);           // g = gelu(conv)*mixed
    conv1x1_wmma<32, 64, 1><<<dim3(P / 64, BATCH), 128, 0, stream>>>(
        gbuf, wdown, down_b, nullptr, tbuf, P);          // down + bias
    conv1x1_wmma<64, 32, 3><<<dim3(P / 64, BATCH), 128, 0, stream>>>(
        tbuf, wup, up_b, attnb, ubuf, P);                // up + bias + attn residual
    conv1x1_wmma<64, 64, 0><<<dim3(P / 64, BATCH), 128, 0, stream>>>(
        ubuf, wproj, nullptr, nullptr, prbuf, P);        // proj
    dw3x3_reflect_k<<<dim3(BATCH * 64 * PIX / 256), 256, 0, stream>>>(prbuf, lp_w, lp_b, lpbuf);
    upsample_k<<<dim3(BATCH * 64 * 256 * 256 / 256), 256, 0, stream>>>(lpbuf, (float*)d_out);
}
